// ProjectedAdaptiveLogSoftmax_66254165508793
// MI455X (gfx1250) — matrix operand order repaired and checked by reading, not compiled, not run on hardware
//
#include <hip/hip_runtime.h>
#include <hip/hip_bf16.h>
#include <math.h>

typedef __bf16 bf16_t;
typedef __attribute__((ext_vector_type(16))) __bf16 v16bf;
typedef __attribute__((ext_vector_type(8)))  __bf16 v8bf;
typedef __attribute__((ext_vector_type(8)))  float  v8f;

#define N_TOK 1024
#define NEG_BIG (-3.0e38f)
#define NSPLIT 8      // column splits per cluster
#define MT 4          // 16-row M tiles per block (64 rows/block)
#define ROWS (MT * 16)

// ---------------------------------------------------------------- converts
__global__ void f32_to_bf16_kernel(const float* __restrict__ in,
                                   bf16_t* __restrict__ out, int count) {
  int i = blockIdx.x * blockDim.x + threadIdx.x;
  if (i < count) out[i] = (bf16_t)in[i];
}

// W f32 [K x N] row-major  ->  Wt bf16 [Npad x KP] row-major (transposed),
// zero-padded in both dims. LDS-tiled so both global accesses are coalesced.
__global__ void convert_weight_T_kernel(const float* __restrict__ W,
                                        bf16_t* __restrict__ Wt,
                                        int K, int N, int KP, int Npad) {
  __shared__ float tile[32][33];
  const int nx = blockIdx.x * 32;
  const int ky = blockIdx.y * 32;
  const int tx = threadIdx.x;   // 0..31
  const int ty = threadIdx.y;   // 0..7
#pragma unroll
  for (int i = 0; i < 4; ++i) {
    int k = ky + ty + i * 8, n = nx + tx;
    tile[ty + i * 8][tx] = (k < K && n < N) ? W[(size_t)k * N + n] : 0.0f;
  }
  __syncthreads();
#pragma unroll
  for (int i = 0; i < 4; ++i) {
    int n = nx + ty + i * 8, k = ky + tx;
    if (n < Npad && k < KP)
      Wt[(size_t)n * KP + k] = (bf16_t)tile[tx][ty + i * 8];
  }
}

// P = hidden @ W_proj  (f32 accumulate), stored bf16 [N_TOK x KP], j>=d zeroed
__global__ void proj_kernel(const float* __restrict__ H,
                            const float* __restrict__ Wp,
                            bf16_t* __restrict__ P, int d, int KP) {
  int id = blockIdx.x * blockDim.x + threadIdx.x;
  int n = id / KP, j = id % KP;
  if (n >= N_TOK) return;
  float acc = 0.0f;
  if (j < d) {
    const float* hrow = H + (size_t)n * 1024;
    for (int k = 0; k < 1024; ++k) acc += hrow[k] * Wp[(size_t)k * d + j];
  }
  P[(size_t)n * KP + j] = (bf16_t)acc;
}

// cluster logits [N_TOK x 3] in f32 (tiny; keep full precision)
__global__ void cluster_kernel(const float* __restrict__ H,
                               const float* __restrict__ Wc,
                               const float* __restrict__ bc,
                               float* __restrict__ cl) {
  int id = blockIdx.x * blockDim.x + threadIdx.x;
  if (id >= N_TOK * 3) return;
  int n = id / 3, j = id % 3;
  float acc = bc[j];
  const float* hrow = H + (size_t)n * 1024;
  for (int k = 0; k < 1024; ++k) acc += hrow[k] * Wc[k * 3 + j];
  cl[id] = acc;
}

// --------------------------------------------------- fused GEMM + logsumexp
// A bf16 [N_TOK x KP], Wt bf16 [Npad x KP] (transposed weights).
// Block = 64 rows (4 M-tiles) x one column split; 8 waves sweep 16-col tiles.
// Per B fragment (two 16B loads/lane) we run MT=4 WMMAs against 4 A fragments
// staged in fragment-major LDS layout [kchunk][mt][row][32] so all per-chunk
// ds_loads are immediate offsets off a single base. Branchless online
// (max,sumexp) per lane, shfl_xor merge in the 16-lane group, LDS merge
// across waves; per-split partial (m,s) written to workspace.
template <int KP>
__global__ __launch_bounds__(256) void adasoft_stream_kernel(
    const bf16_t* __restrict__ A, const bf16_t* __restrict__ Wt,
    const float* __restrict__ bias,
    int N, int ntiles, int tilesPerSplit,
    float* __restrict__ pm_out,   // [NSPLIT][N_TOK] partial max
    float* __restrict__ ps_out) { // [NSPLIT][N_TOK] partial sumexp
  extern __shared__ __align__(16) bf16_t As[];   // [KP/32][MT][16][32]
  __shared__ float red_m[8][ROWS];
  __shared__ float red_s[8][ROWS];

  const int tid  = threadIdx.x;
  const int wave = tid >> 5;
  const int lane = tid & 31;
  const int nl   = lane & 15;   // A row within tile AND B column within tile
  const int half = lane >> 4;
  const int rowbase = blockIdx.x * ROWS;
  const int split   = blockIdx.y;
  const int tile0   = split * tilesPerSplit;
  const int tile1   = min(ntiles, tile0 + tilesPerSplit);

  { // stage A into fragment-major layout: As[((ch*MT + mt)*16 + row)*32 + kk]
    const bf16_t* Ag = A + (size_t)rowbase * KP;
    for (int i = tid * 8; i < ROWS * KP; i += 256 * 8) {
      int ch  = i >> 11;          // / (MT*16*32)
      int r2  = i & 2047;
      int mt  = r2 >> 9;          // / (16*32)
      int r3  = r2 & 511;
      int row = r3 >> 5;
      int kk  = r3 & 31;
      *(v8bf*)(&As[i]) =
          *(const v8bf*)(&Ag[(size_t)(mt * 16 + row) * KP + ch * 32 + kk]);
    }
  }
  __syncthreads();

  const int abase = nl * 32 + half * 8;  // lane's element offset within block

  float rm[MT * 8], rs[MT * 8];
#pragma unroll
  for (int v = 0; v < MT * 8; ++v) { rm[v] = NEG_BIG; rs[v] = 0.0f; }

  for (int tile = tile0 + wave; tile < tile1; tile += 8) {
    const int n = (tile << 4) + nl;                 // column within cluster
    const bf16_t* Bp = Wt + (size_t)n * KP;
    __builtin_prefetch(Bp + (size_t)128 * KP, 0, 1); // this wave's next tile
    v8f c[MT];
#pragma unroll
    for (int mt = 0; mt < MT; ++mt) c[mt] = (v8f){};
#pragma unroll 2
    for (int ch = 0; ch < KP / 32; ++ch) {
      // B 16-bit 32x16 layout: lane n=nl, K runs 8h..8h+7 and 16+8h..16+8h+7
      v8bf blo = *(const v8bf*)(Bp + ch * 32 + half * 8);
      v8bf bhi = *(const v8bf*)(Bp + ch * 32 + 16 + half * 8);
      v16bf b = __builtin_shufflevector(blo, bhi, 0, 1, 2, 3, 4, 5, 6, 7,
                                        8, 9, 10, 11, 12, 13, 14, 15);
      v16bf afr[MT];
#pragma unroll
      for (int mt = 0; mt < MT; ++mt) {
        const bf16_t* ap = As + ch * (MT * 512) + mt * 512 + abase;
        v8bf alo = *(const v8bf*)(ap);
        v8bf ahi = *(const v8bf*)(ap + 16);
        afr[mt] = __builtin_shufflevector(alo, ahi, 0, 1, 2, 3, 4, 5, 6, 7,
                                          8, 9, 10, 11, 12, 13, 14, 15);
      }
#pragma unroll
      for (int mt = 0; mt < MT; ++mt)
        c[mt] = __builtin_amdgcn_wmma_f32_16x16x32_bf16(false, afr[mt], false,
                                                        b, (short)0, c[mt],
                                                        false, false);
    }
    if (n < N) {
      float bn = bias[n];
#pragma unroll
      for (int mt = 0; mt < MT; ++mt) {
#pragma unroll
        for (int v = 0; v < 8; ++v) {
          float x = c[mt][v] + bn;             // row mt*16+half*8+v, col n
          int idx = mt * 8 + v;
          float mo = rm[idx];
          float M = fmaxf(mo, x);              // branchless online update
          rs[idx] = rs[idx] * __expf(mo - M) + __expf(x - M);
          rm[idx] = M;
        }
      }
    }
  }

  // merge (m,s) across the 16 columns of this lane group
#pragma unroll
  for (int v = 0; v < MT * 8; ++v) {
#pragma unroll
    for (int s = 1; s < 16; s <<= 1) {
      float om = __shfl_xor(rm[v], s, 32);
      float os = __shfl_xor(rs[v], s, 32);
      float M = fmaxf(rm[v], om);
      rs[v] = rs[v] * __expf(rm[v] - M) + os * __expf(om - M);
      rm[v] = M;
    }
  }
  if (nl == 0) {
#pragma unroll
    for (int mt = 0; mt < MT; ++mt)
#pragma unroll
      for (int v = 0; v < 8; ++v) {
        red_m[wave][mt * 16 + half * 8 + v] = rm[mt * 8 + v];
        red_s[wave][mt * 16 + half * 8 + v] = rs[mt * 8 + v];
      }
  }
  __syncthreads();
  if (tid < ROWS) {
    float M = NEG_BIG;
    for (int w = 0; w < 8; ++w) M = fmaxf(M, red_m[w][tid]);
    float S = 0.0f;
    for (int w = 0; w < 8; ++w) S += red_s[w][tid] * __expf(red_m[w][tid] - M);
    pm_out[(size_t)split * N_TOK + rowbase + tid] = M;
    ps_out[(size_t)split * N_TOK + rowbase + tid] = S;
  }
}

// ------------------------------------------- target logit gather (1 wave/tok)
__global__ void gather_kernel(
    const bf16_t* __restrict__ A0, const bf16_t* __restrict__ P1,
    const bf16_t* __restrict__ P2, const bf16_t* __restrict__ P3,
    const bf16_t* __restrict__ WtH, const bf16_t* __restrict__ Wt1,
    const bf16_t* __restrict__ Wt2, const bf16_t* __restrict__ Wt3,
    const float* __restrict__ bh, const float* __restrict__ b1,
    const float* __restrict__ b2, const float* __restrict__ b3,
    const int* __restrict__ target, float* __restrict__ gath) {
  int gid = blockIdx.x * blockDim.x + threadIdx.x;
  int n = gid >> 5, lane = gid & 31;
  if (n >= N_TOK) return;
  int t = target[n];            // uniform within the wave
  const bf16_t *Ap, *Wp; const float* bp; int KP, c, col;
  if (t < 20000)       { c = 0; col = t;          Ap = A0 + (size_t)n * 1024; Wp = WtH; bp = bh; KP = 1024; }
  else if (t < 60000)  { c = 1; col = t - 20000;  Ap = P1 + (size_t)n * 256;  Wp = Wt1; bp = b1; KP = 256; }
  else if (t < 180000) { c = 2; col = t - 60000;  Ap = P2 + (size_t)n * 64;   Wp = Wt2; bp = b2; KP = 64;  }
  else                 { c = 3; col = t - 180000; Ap = P3 + (size_t)n * 32;   Wp = Wt3; bp = b3; KP = 32;  }
  const bf16_t* Wc = Wp + (size_t)col * KP;
  float acc = 0.0f;
  for (int k = lane; k < KP; k += 32) acc += (float)Ap[k] * (float)Wc[k];
#pragma unroll
  for (int s = 16; s; s >>= 1) acc += __shfl_xor(acc, s, 32);
  if (lane == 0) gath[c * N_TOK + n] = acc + bp[col];
}

// ----------------------------------------------- merge split partials -> lse
__global__ void merge_lse_kernel(const float* __restrict__ pm,
                                 const float* __restrict__ ps,
                                 float* __restrict__ lse) {
  int id = blockIdx.x * blockDim.x + threadIdx.x;  // over 4 * N_TOK
  if (id >= 4 * N_TOK) return;
  int c = id >> 10, r = id & (N_TOK - 1);
  const float* pmc = pm + (size_t)c * NSPLIT * N_TOK + r;
  const float* psc = ps + (size_t)c * NSPLIT * N_TOK + r;
  float M = NEG_BIG;
#pragma unroll
  for (int s = 0; s < NSPLIT; ++s) M = fmaxf(M, pmc[(size_t)s * N_TOK]);
  float S = 0.0f;
#pragma unroll
  for (int s = 0; s < NSPLIT; ++s)
    S += psc[(size_t)s * N_TOK] * __expf(pmc[(size_t)s * N_TOK] - M);
  lse[id] = M + __logf(S);
}

// ------------------------------------------------------------- final NLL
__global__ void combine_kernel(const int* __restrict__ target,
                               const float* __restrict__ cl,
                               const float* __restrict__ lse,
                               const float* __restrict__ gath,
                               float* __restrict__ out) {
  int n = blockIdx.x * blockDim.x + threadIdx.x;
  if (n >= N_TOK) return;
  int t = target[n];
  float c0 = cl[n * 3 + 0], c1 = cl[n * 3 + 1], c2 = cl[n * 3 + 2];
  float lh = lse[n];  // logsumexp over the 20000 head columns
  float M = fmaxf(fmaxf(lh, c0), fmaxf(c1, c2));
  float S = __expf(lh - M) + __expf(c0 - M) + __expf(c1 - M) + __expf(c2 - M);
  float LSE = M + __logf(S);
  float lp;
  if (t < 20000) {
    lp = gath[n] - LSE;
  } else {
    int i = (t < 60000) ? 1 : ((t < 180000) ? 2 : 3);
    float clog = (i == 1) ? c2 : ((i == 2) ? c1 : c0);  // head_lp[:, -i]
    lp = (clog - LSE) + (gath[i * N_TOK + n] - lse[i * N_TOK + n]);
  }
  out[n] = -lp;
}

// ------------------------------------------------------------------ launch
extern "C" void kernel_launch(void* const* d_in, const int* in_sizes, int n_in,
                              void* d_out, int out_size, void* d_ws,
                              size_t ws_size, hipStream_t stream) {
  const float* hidden   = (const float*)d_in[0];
  const int*   target   = (const int*)d_in[1];
  const float* W_head   = (const float*)d_in[2];
  const float* b_head   = (const float*)d_in[3];
  const float* W_clu    = (const float*)d_in[4];
  const float* b_clu    = (const float*)d_in[5];
  const float* W_proj1  = (const float*)d_in[6];
  const float* W_tail1  = (const float*)d_in[7];
  const float* b_tail1  = (const float*)d_in[8];
  const float* W_proj2  = (const float*)d_in[9];
  const float* W_tail2  = (const float*)d_in[10];
  const float* b_tail2  = (const float*)d_in[11];
  const float* W_proj3  = (const float*)d_in[12];
  const float* W_tail3  = (const float*)d_in[13];
  const float* b_tail3  = (const float*)d_in[14];
  float* out = (float*)d_out;

  char* ws = (char*)d_ws;
  size_t off = 0;
  auto alloc = [&](size_t bytes) -> char* {
    char* p = ws + off;
    off = (off + bytes + 255) & ~(size_t)255;
    return p;
  };
  bf16_t* A0  = (bf16_t*)alloc((size_t)N_TOK * 1024 * 2);
  bf16_t* P1  = (bf16_t*)alloc((size_t)N_TOK * 256 * 2);
  bf16_t* P2  = (bf16_t*)alloc((size_t)N_TOK * 64 * 2);
  bf16_t* P3  = (bf16_t*)alloc((size_t)N_TOK * 32 * 2);
  bf16_t* WtH = (bf16_t*)alloc((size_t)20000 * 1024 * 2);
  bf16_t* Wt1 = (bf16_t*)alloc((size_t)40000 * 256 * 2);
  bf16_t* Wt2 = (bf16_t*)alloc((size_t)120000 * 64 * 2);
  bf16_t* Wt3 = (bf16_t*)alloc((size_t)87744 * 32 * 2);
  float*  CL  = (float*)alloc((size_t)N_TOK * 3 * 4);
  float*  PM  = (float*)alloc((size_t)4 * NSPLIT * N_TOK * 4);
  float*  PS  = (float*)alloc((size_t)4 * NSPLIT * N_TOK * 4);
  float*  LSE = (float*)alloc((size_t)4 * N_TOK * 4);
  float*  GTH = (float*)alloc((size_t)4 * N_TOK * 4);

  // 1) bf16 conversions / transposes
  f32_to_bf16_kernel<<<(1024 * 1024) / 256, 256, 0, stream>>>(hidden, A0,
                                                              1024 * 1024);
  dim3 tb(32, 8);
  convert_weight_T_kernel<<<dim3(20000 / 32, 1024 / 32), tb, 0, stream>>>(
      W_head, WtH, 1024, 20000, 1024, 20000);
  convert_weight_T_kernel<<<dim3(40000 / 32, 256 / 32), tb, 0, stream>>>(
      W_tail1, Wt1, 256, 40000, 256, 40000);
  convert_weight_T_kernel<<<dim3(120000 / 32, 64 / 32), tb, 0, stream>>>(
      W_tail2, Wt2, 64, 120000, 64, 120000);
  convert_weight_T_kernel<<<dim3(87744 / 32, 1), tb, 0, stream>>>(
      W_tail3, Wt3, 16, 87735, 32, 87744);

  // 2) projections (f32 accumulate -> bf16) and cluster logits
  proj_kernel<<<(N_TOK * 256) / 256, 256, 0, stream>>>(hidden, W_proj1, P1,
                                                       256, 256);
  proj_kernel<<<(N_TOK * 64) / 256, 256, 0, stream>>>(hidden, W_proj2, P2, 64,
                                                      64);
  proj_kernel<<<(N_TOK * 32) / 256, 256, 0, stream>>>(hidden, W_proj3, P3, 16,
                                                      32);
  cluster_kernel<<<(N_TOK * 3 + 255) / 256, 256, 0, stream>>>(hidden, W_clu,
                                                              b_clu, CL);

  // 3) fused WMMA GEMM + online logsumexp, per cluster.
  //    Grid: (row blocks of 64) x (NSPLIT column splits).
  dim3 grid(N_TOK / ROWS, NSPLIT);
  auto tps = [](int ntiles) { return (ntiles + NSPLIT - 1) / NSPLIT; };
  {
    int nt = 20000 / 16;
    adasoft_stream_kernel<1024><<<grid, 256, ROWS * 1024 * 2, stream>>>(
        A0, WtH, b_head, 20000, nt, tps(nt),
        PM + 0 * NSPLIT * N_TOK, PS + 0 * NSPLIT * N_TOK);
  }
  {
    int nt = 40000 / 16;
    adasoft_stream_kernel<256><<<grid, 256, ROWS * 256 * 2, stream>>>(
        P1, Wt1, b_tail1, 40000, nt, tps(nt),
        PM + 1 * NSPLIT * N_TOK, PS + 1 * NSPLIT * N_TOK);
  }
  {
    int nt = 120000 / 16;
    adasoft_stream_kernel<64><<<grid, 256, ROWS * 64 * 2, stream>>>(
        P2, Wt2, b_tail2, 120000, nt, tps(nt),
        PM + 2 * NSPLIT * N_TOK, PS + 2 * NSPLIT * N_TOK);
  }
  {
    int nt = 87744 / 16;
    adasoft_stream_kernel<32><<<grid, 256, ROWS * 32 * 2, stream>>>(
        P3, Wt3, b_tail3, 87735, nt, tps(nt),
        PM + 3 * NSPLIT * N_TOK, PS + 3 * NSPLIT * N_TOK);
  }

  // 3b) gathered target logits (one wave per token)
  gather_kernel<<<(N_TOK * 32) / 256, 256, 0, stream>>>(
      A0, P1, P2, P3, WtH, Wt1, Wt2, Wt3, b_head, b_tail1, b_tail2, b_tail3,
      target, GTH);

  // 4) merge split partials into per-row logsumexp, then per-token NLL
  merge_lse_kernel<<<(4 * N_TOK) / 256, 256, 0, stream>>>(PM, PS, LSE);
  combine_kernel<<<(N_TOK + 255) / 256, 256, 0, stream>>>(target, CL, LSE, GTH,
                                                          out);
}
